// Model_nefnet_54958401519914
// MI455X (gfx1250) — matrix operand order repaired and checked by reading, not compiled, not run on hardware
//
#include <hip/hip_runtime.h>
#include <math.h>

// MI455X / gfx1250, wave32. Grouped conv3 layers = implicit GEMM on
// v_wmma_f32_16x16x32_f16 (f32 accumulate).
//  - Activations: f16, layout [B, T+2, C] (zeroed halo rows at t=-1 and t=T)
//    so the conv inner loop has NO bounds checks / branches at all.
//  - B-fragment: 1 contiguous 32B load per lane; C/D tile: 16B store per lane.
//  - Weights: packed once into exact A-fragment order -> 32B load per lane.
//  - NT templated strip (16co x NT*16 t) so all inner loops fully unroll.

typedef _Float16 h16;
typedef __attribute__((ext_vector_type(16))) _Float16 v16h;
typedef __attribute__((ext_vector_type(8)))  _Float16 v8h;
typedef __attribute__((ext_vector_type(8)))  float    v8f;

// ---------------------------------------------------------------------------
// Weight pack: w [CoutTot, CinPG, taps] (f32) -> fragment order f16:
//   out[(((tile*chunks + chunk)*taps + tap)*32 + lane)*16 + e]
// A 16x32 f16 packing used here (contiguous k, same as B loads): lane = 16*half + M ; K = e + 16*half
// ---------------------------------------------------------------------------
__global__ void pack_w_k(const float* __restrict__ w, h16* __restrict__ out,
                         int CinPG, int taps, int total)
{
    int idx = blockIdx.x * blockDim.x + threadIdx.x;
    if (idx >= total) return;
    int e    = idx & 15;
    int lane = (idx >> 4) & 31;
    int rest = idx >> 9;
    int tap  = rest % taps;
    int ck   = rest / taps;              // tile*chunks + chunk
    int chunks = CinPG >> 5;
    int tile  = ck / chunks;
    int chunk = ck - tile * chunks;
    int half = lane >> 4, l15 = lane & 15;
    int kA = e + (half << 4);
    int co = tile * 16 + l15;
    int ci = (chunk << 5) + kA;
    out[idx] = (h16)w[((size_t)co * CinPG + ci) * taps + tap];
}

// zero halo rows 0 and T+1 of buf [64, T+2, C]
__global__ void halo_zero(h16* __restrict__ buf, int T, int C)
{
    int idx = blockIdx.x * blockDim.x + threadIdx.x;
    if (idx >= 64 * 2 * C) return;
    int c = idx % C;
    int s = (idx / C) & 1;
    int b = idx / (2 * C);
    buf[((size_t)b * (T + 2) + (s ? (T + 1) : 0)) * C + c] = (h16)0.f;
}

// ---------------------------------------------------------------------------
// Grouped conv1d k=3 pad=1, implicit GEMM on WMMA. Branch-free inner loop.
// in  : f16 [B, T+2, CinTot] halo-padded ; out : f16 [B, T+2, CoutTot] (interior)
// resmode: 0 none; 1 identity (resin [B,T+2,CoutTot]); 2 1x1-conv residual
// bng/cb/bnbe: eval-BN fold: v = (v + cb[c]) * bng[c]/sqrt(1+1e-5) + bnbe[c]
// ---------------------------------------------------------------------------
template <int NT>
__global__ __launch_bounds__(32)
void conv3_wmma_t(const h16* __restrict__ in, const h16* __restrict__ wp,
                  const h16* __restrict__ resin, const h16* __restrict__ rwp,
                  const float* __restrict__ rb,
                  const float* __restrict__ bng, const float* __restrict__ cb,
                  const float* __restrict__ bnbe,
                  h16* __restrict__ out,
                  int CinPG, int CinRPG, int CoutPG, int T, int groups,
                  int resmode, int dorelu)
{
    const int lane = threadIdx.x;
    const int half = lane >> 4;
    const int l15  = lane & 15;
    const int tile = blockIdx.x;
    const int coBase = tile * 16;
    const int t0 = blockIdx.y * (NT * 16);
    const int b  = blockIdx.z;
    const int grp = coBase / CoutPG;
    const int CinTot  = CinPG * groups;
    const int CoutTot = CoutPG * groups;
    const int ciBase = grp * CinPG;
    const int chunks = CinPG >> 5;
    const int ldI = T + 2;

    v8f acc[NT] = {};

    const h16* wtile = wp + ((size_t)tile * chunks * 3) * 512 + (size_t)lane * 16;

    for (int c = 0; c < chunks; ++c) {
        // base at padded row 0 (= logical t-1); row index below is t0+tt*16+l15+tap
        const h16* inc = in + (size_t)b * ldI * CinTot + ciBase + (c << 5) + (half << 4);
        #pragma unroll
        for (int tap = 0; tap < 3; ++tap) {
            const v16h a = *(const v16h*)(wtile + (size_t)(c * 3 + tap) * 512);
            #pragma unroll
            for (int tt = 0; tt < NT; ++tt) {
                const int row = t0 + tt * 16 + l15 + tap;   // in [0, T+1], always valid
                const v16h bf = *(const v16h*)(inc + (size_t)row * CinTot);
                acc[tt] = __builtin_amdgcn_wmma_f32_16x16x32_f16(
                              false, a, false, bf, (short)0, acc[tt], false, false);
            }
        }
    }

    if (resmode == 2) {   // fold 1x1 residual conv into the reduction (interior rows)
        const int CinRTot = CinRPG * groups;
        const int rchunks = CinRPG >> 5;
        const h16* rwtile = rwp + ((size_t)tile * rchunks) * 512 + (size_t)lane * 16;
        for (int c = 0; c < rchunks; ++c) {
            const v16h a = *(const v16h*)(rwtile + (size_t)c * 512);
            const h16* inc = resin + ((size_t)b * ldI + 1) * CinRTot
                           + grp * CinRPG + (c << 5) + (half << 4);
            #pragma unroll
            for (int tt = 0; tt < NT; ++tt) {
                const int t = t0 + tt * 16 + l15;
                const v16h bf = *(const v16h*)(inc + (size_t)t * CinRTot);
                acc[tt] = __builtin_amdgcn_wmma_f32_16x16x32_f16(
                              false, a, false, bf, (short)0, acc[tt], false, false);
            }
        }
    }

    // Epilogue. C/D f32 layout: VGPR r -> M = r + 8*half, N = l15.
    const int coE0 = coBase + (half << 3);
    #pragma unroll
    for (int tt = 0; tt < NT; ++tt) {
        const int t = t0 + tt * 16 + l15;
        const size_t obase = ((size_t)b * ldI + 1 + t) * CoutTot + coE0;
        v8h rv = {};
        if (resmode == 1) rv = *(const v8h*)(resin + obase);
        v8h ov;
        #pragma unroll
        for (int r = 0; r < 8; ++r) {
            float v = acc[tt][r];
            const int coE = coE0 + r;
            if (bng) v = (v + cb[coE]) * (bng[coE] * 0.9999950000374997f) + bnbe[coE];
            if (resmode == 1)      v += (float)rv[r];
            else if (resmode == 2) v += rb[coE];
            if (dorelu) v = fmaxf(v, 0.f);
            ov[r] = (h16)v;
        }
        *(v8h*)(out + obase) = ov;
    }
}

// ---------------------------------------------------------------------------
// Encoder conv (groups=4, Cin=1/grp, k=7, s=2, pad=3) + ReLU + maxpool k3 s2 p1.
// x f32 [64,4,512] -> out f16 [64, 130, 256] (interior rows 1..128)
// ---------------------------------------------------------------------------
__global__ void enc_conv7_pool_t(const float* __restrict__ x, const float* __restrict__ w,
                                 h16* __restrict__ out)
{
    int idx = blockIdx.x * blockDim.x + threadIdx.x;
    if (idx >= 64*128*256) return;
    int co = idx & 255;
    int t  = (idx >> 8) & 127;
    int b  = idx >> 15;
    int g  = co >> 6;
    const float* wr = w + co * 7;
    const float* xr = x + ((size_t)b * 4 + g) * 512;
    float m = 0.f;                      // relu floor
    for (int d = -1; d <= 1; ++d) {
        int u = 2*t + d;
        if (u < 0 || u > 255) continue;
        float acc = 0.f;
        int base = 2*u - 3;
        #pragma unroll
        for (int k = 0; k < 7; ++k) {
            int s = base + k;
            if (s >= 0 && s < 512) acc += wr[k] * xr[s];
        }
        m = fmaxf(m, fmaxf(acc, 0.f));
    }
    out[((size_t)b*130 + 1 + t)*256 + co] = (h16)m;
}

// theta_enc (12 feats) @ W[12,outF] + b  -> f32
__global__ void theta_mlp(const float* __restrict__ theta, const float* __restrict__ w,
                          const float* __restrict__ bias, float* __restrict__ out,
                          int rows, int outF)
{
    int idx = blockIdx.x * blockDim.x + threadIdx.x;
    if (idx >= rows * outF) return;
    int f = idx % outF, r = idx / outF;
    float t0 = theta[r*2], t1 = theta[r*2+1];
    float e[12];
    e[0]=sinf(t0); e[1]=sinf(t1); e[2]=cosf(t0); e[3]=cosf(t1);
    #pragma unroll
    for (int i = 0; i < 4; ++i) { e[4+i]=sinf(e[i]); e[8+i]=cosf(e[i]); }
    float acc = bias[f];
    #pragma unroll
    for (int i = 0; i < 12; ++i) acc += e[i] * w[i*outF + f];
    out[idx] = acc;
}

// ew[b,t,c] = th[b,c] * w[b,t,c]   over padded [64,130,512]
__global__ void ew_mul_t(const float* __restrict__ th, const h16* __restrict__ w,
                         h16* __restrict__ out)
{
    int idx = blockIdx.x * blockDim.x + threadIdx.x;
    if (idx >= 64*128*512) return;
    int c = idx & 511;
    int t = (idx >> 9) & 127;
    int b = idx >> 16;
    size_t p = ((size_t)b*130 + 1 + t)*512 + c;
    out[p] = (h16)(th[b*512 + c] * (float)w[p]);
}

// split ew2 [64,130,512] -> z1/z2 [64,130,256] (per lead: 128ch -> 64+64)
__global__ void split_t(const h16* __restrict__ ew2, h16* __restrict__ z1,
                        h16* __restrict__ z2)
{
    int idx = blockIdx.x * blockDim.x + threadIdx.x;
    if (idx >= 64*128*256) return;
    int c = idx & 255;
    int t = (idx >> 8) & 127;
    int b = idx >> 15;
    int l = c >> 6, cc = c & 63;
    size_t row = (size_t)b*130 + 1 + t;
    size_t src = row * 512 + l*128 + cc;
    size_t dst = row * 256 + c;
    z1[dst] = ew2[src];
    z2[dst] = ew2[src + 64];
}

// roi_align: x [64,130,512] -> out [64,18,3584] (c' = c*7 + seg); scale 0.25
__global__ void roi_align_t(const h16* __restrict__ x, const float* __restrict__ rois,
                            h16* __restrict__ out)
{
    int idx = blockIdx.x * blockDim.x + threadIdx.x;
    if (idx >= 64*16*3584) return;
    int cc = idx % 3584;
    int j  = (idx / 3584) & 15;
    int b  = idx / (16*3584);
    int c = cc / 7, r = cc % 7;
    float pts[8]; pts[0]=0.f; pts[7]=128.f;
    #pragma unroll
    for (int i = 0; i < 6; ++i) pts[i+1] = rois[b*6+i] * 0.25f;
    float s = pts[r], e = pts[r+1];
    float pos = s + (e - s) * ((j + 0.5f) / 16.f) - 0.5f;
    pos = fminf(fmaxf(pos, 0.f), 127.f);
    int i0 = (int)floorf(pos);
    int i1 = (i0+1 < 128) ? i0+1 : 127;
    float f = pos - (float)i0;
    const h16* xb = x + ((size_t)b*130 + 1)*512 + c;
    float v = (float)xb[(size_t)i0*512]*(1.f-f) + (float)xb[(size_t)i1*512]*f;
    out[((size_t)b*18 + 1 + j)*3584 + cc] = (h16)v;
}

// grouped ConvTranspose1d k=2 s=2, groups=28:
// x [64,18,3584], w f32 [3584,64,2] -> out [64,34,1792] (+bias)
__global__ void convT_t(const h16* __restrict__ x, const float* __restrict__ w,
                        const float* __restrict__ bias, h16* __restrict__ out)
{
    int idx = blockIdx.x * blockDim.x + threadIdx.x;
    if (idx >= 64*32*1792) return;
    int gc = idx % 1792;
    int u  = (idx / 1792) & 31;
    int b  = idx / (32*1792);
    int g = gc >> 6, o = gc & 63;
    int t = u >> 1, k = u & 1;
    const h16* xr = x + ((size_t)b*18 + 1 + t)*3584 + g*128;
    const float* wr = w + (size_t)g*128*128 + o*2 + k;
    float acc = bias[gc];
    for (int i = 0; i < 128; ++i)
        acc += (float)xr[i] * wr[(size_t)i*128];
    out[((size_t)b*34 + 1 + u)*1792 + gc] = (h16)acc;
}

// roi_pooling_reverse: x [64,34,3584] (c' = c*7+seg) -> out [64,130,512]
__global__ void roi_rev_t(const h16* __restrict__ x, const float* __restrict__ rois,
                          h16* __restrict__ out)
{
    int idx = blockIdx.x * blockDim.x + threadIdx.x;
    if (idx >= 64*128*512) return;
    int c = idx & 511;
    int t = (idx >> 9) & 127;
    int b = idx >> 16;
    float pts[8]; pts[0]=0.f; pts[7]=128.f;
    #pragma unroll
    for (int i = 0; i < 6; ++i) pts[i+1] = rois[b*6+i]*0.25f;
    float tc = t + 0.5f;
    int seg = 0;
    #pragma unroll
    for (int i = 0; i < 6; ++i) seg += (pts[i+1] <= tc) ? 1 : 0;
    float s = pts[seg], e = pts[seg+1];
    float u = (tc - s) / fmaxf(e - s, 1e-6f) * 32.f - 0.5f;
    u = fminf(fmaxf(u, 0.f), 31.f);
    int i0 = (int)floorf(u);
    int i1 = (i0+1 < 32) ? i0+1 : 31;
    float f = u - (float)i0;
    const h16* xb = x + ((size_t)b*34 + 1)*3584 + c*7 + seg;
    float v = (float)xb[(size_t)i0*3584]*(1.f-f) + (float)xb[(size_t)i1*3584]*f;
    out[((size_t)b*130 + 1 + t)*512 + c] = (h16)v;
}

// mean over 4 leads: x [64,130,512] -> out [64,128,128] (unpadded)
__global__ void lead_mean_t(const h16* __restrict__ x, h16* __restrict__ out)
{
    int idx = blockIdx.x * blockDim.x + threadIdx.x;
    if (idx >= 64*128*128) return;
    int c = idx & 127;
    int t = (idx >> 7) & 127;
    int b = idx >> 14;
    const h16* xr = x + ((size_t)b*130 + 1 + t)*512 + c;
    float a = 0.f;
    #pragma unroll
    for (int l = 0; l < 4; ++l) a += (float)xr[l*128];
    out[idx] = (h16)(a * 0.25f);
}

// build latent / sh_p / sh_l and multiply by qt: out [64,130,256] (interior)
__global__ void build_qz_t(const h16* __restrict__ z1, const h16* __restrict__ z2,
                           const h16* __restrict__ z1m, const h16* __restrict__ z2m,
                           const float* __restrict__ qt, h16* __restrict__ out, int mode)
{
    int idx = blockIdx.x * blockDim.x + threadIdx.x;
    if (idx >= 64*128*256) return;
    int c = idx & 255;
    int t = (idx >> 8) & 127;
    int b = idx >> 15;
    size_t prow = (size_t)b*130 + 1 + t;   // padded row (z1/z2/out)
    size_t urow = (size_t)b*128 + t;       // unpadded row (z1m/z2m)
    float v;
    if (c < 128)
        v = (mode == 1) ? (float)z1[prow*512 + c]          // lead 0 of z1
                        : (float)z1m[urow*128 + c];
    else {
        int cc = c - 128;
        v = (mode == 2) ? (float)z2[prow*512 + cc]         // lead 0 of z2
                        : (float)z2m[urow*128 + cc];
    }
    out[prow*256 + c] = (h16)(qt[b*256 + c] * v);
}

// linear 2x upsample (align_corners=False): x [64,T+2,C] -> out [64,2T+2,C]
__global__ void up2_t(const h16* __restrict__ x, h16* __restrict__ out, int C, int T)
{
    int idx = blockIdx.x * blockDim.x + threadIdx.x;
    if (idx >= 64*C*2*T) return;
    int c = idx % C;
    int i = (idx / C) % (2*T);
    int b = idx / (C*2*T);
    int t = i >> 1;
    const h16* xb = x + ((size_t)b*(T+2) + 1)*C + c;
    float v;
    if ((i & 1) == 0)
        v = (t == 0)   ? (float)xb[0]
                       : 0.25f*(float)xb[(size_t)(t-1)*C] + 0.75f*(float)xb[(size_t)t*C];
    else
        v = (t == T-1) ? (float)xb[(size_t)(T-1)*C]
                       : 0.75f*(float)xb[(size_t)t*C] + 0.25f*(float)xb[(size_t)(t+1)*C];
    out[((size_t)b*(2*T+2) + 1 + i)*C + c] = (h16)v;
}

// final conv (1,64,3)+bias then sigmoid(v/3): x [64,514,64] (halo-zeroed) -> f32 [64,512]
__global__ void outconv_sig_t(const h16* __restrict__ x, const float* __restrict__ w,
                              const float* __restrict__ bias, float* __restrict__ out)
{
    int idx = blockIdx.x * blockDim.x + threadIdx.x;
    if (idx >= 64*512) return;
    int t = idx & 511;
    int b = idx >> 9;
    float acc = bias[0];
    #pragma unroll
    for (int k = 0; k < 3; ++k) {
        const h16* xr = x + ((size_t)b*514 + t + k)*64;   // halo rows are zero
        for (int ci = 0; ci < 64; ++ci)
            acc += w[ci*3 + k] * (float)xr[ci];
    }
    float v = acc * (1.f/3.f);
    out[idx] = 1.f / (1.f + expf(-v));
}

// ---------------------------------------------------------------------------
static inline dim3 g1(long long n, int bs = 256) { return dim3((unsigned)((n + bs - 1) / bs)); }

static void launch_conv(const h16* in, const h16* wp, const h16* resin, const h16* rwp,
                        const float* rb, const float* bng, const float* cb,
                        const float* bnbe, h16* out,
                        int CinPG, int CinRPG, int CoutPG, int T, int groups,
                        int resmode, int relu, hipStream_t s)
{
    // zero the input halo rows, then run the branch-free WMMA conv
    halo_zero<<<g1(64LL*2*CinPG*groups), 256, 0, s>>>((h16*)in, T, CinPG*groups);
    if (T >= 64) {
        dim3 grid(CoutPG*groups/16, T/64, 64);
        conv3_wmma_t<4><<<grid, 32, 0, s>>>(in, wp, resin, rwp, rb, bng, cb, bnbe, out,
                                            CinPG, CinRPG, CoutPG, T, groups, resmode, relu);
    } else if (T == 32) {
        dim3 grid(CoutPG*groups/16, 1, 64);
        conv3_wmma_t<2><<<grid, 32, 0, s>>>(in, wp, resin, rwp, rb, bng, cb, bnbe, out,
                                            CinPG, CinRPG, CoutPG, T, groups, resmode, relu);
    } else {
        dim3 grid(CoutPG*groups/16, 1, 64);
        conv3_wmma_t<1><<<grid, 32, 0, s>>>(in, wp, resin, rwp, rb, bng, cb, bnbe, out,
                                            CinPG, CinRPG, CoutPG, T, groups, resmode, relu);
    }
}

extern "C" void kernel_launch(void* const* d_in, const int* in_sizes, int n_in,
                              void* d_out, int out_size, void* d_ws, size_t ws_size,
                              hipStream_t stream)
{
    (void)in_sizes; (void)n_in; (void)out_size; (void)ws_size;
    // Positional inputs, then params flattened in JAX pytree (alphabetical) order.
    const float* x       = (const float*)d_in[0];
    const float* ithetas = (const float*)d_in[1];
    const float* qtheta  = (const float*)d_in[2];
    const float* rois    = (const float*)d_in[3];
    const float* ct_b    = (const float*)d_in[4];
    const float* ct_w    = (const float*)d_in[5];
    const float* dc1_b1  = (const float*)d_in[6];
    const float* dc1_b2  = (const float*)d_in[7];
    const float* dc1_be1 = (const float*)d_in[8];
    const float* dc1_be2 = (const float*)d_in[9];
    const float* dc1_g1  = (const float*)d_in[10];
    const float* dc1_g2  = (const float*)d_in[11];
    const float* dc1_w1  = (const float*)d_in[12];
    const float* dc1_w2  = (const float*)d_in[13];
    const float* dc2_b1  = (const float*)d_in[14];
    const float* dc2_b2  = (const float*)d_in[15];
    const float* dc2_be1 = (const float*)d_in[16];
    const float* dc2_be2 = (const float*)d_in[17];
    const float* dc2_g1  = (const float*)d_in[18];
    const float* dc2_g2  = (const float*)d_in[19];
    const float* dc2_w1  = (const float*)d_in[20];
    const float* dc2_w2  = (const float*)d_in[21];
    const float* eb1_c1  = (const float*)d_in[22];
    const float* eb1_c2  = (const float*)d_in[23];
    const float* eb1_rb  = (const float*)d_in[24];
    const float* eb1_rw  = (const float*)d_in[25];
    const float* eb2_c1  = (const float*)d_in[26];
    const float* eb2_c2  = (const float*)d_in[27];
    const float* enc_c1  = (const float*)d_in[28];
    const float* mlp1_b  = (const float*)d_in[29];
    const float* mlp1_w  = (const float*)d_in[30];
    const float* mlp2_b  = (const float*)d_in[31];
    const float* mlp2_w  = (const float*)d_in[32];
    const float* out_b   = (const float*)d_in[33];
    const float* out_w   = (const float*)d_in[34];
    const float* wc_c1   = (const float*)d_in[35];
    const float* wc_c2   = (const float*)d_in[36];
    const float* z1_c1   = (const float*)d_in[37];
    const float* z1_c2   = (const float*)d_in[38];
    const float* z1_rb   = (const float*)d_in[39];
    const float* z1_rw   = (const float*)d_in[40];
    const float* z2_c1   = (const float*)d_in[41];
    const float* z2_c2   = (const float*)d_in[42];
    const float* z2_rb   = (const float*)d_in[43];
    const float* z2_rw   = (const float*)d_in[44];
    const float* z2b1_c1 = (const float*)d_in[45];
    const float* z2b1_c2 = (const float*)d_in[46];
    const float* z2b2_c1 = (const float*)d_in[47];
    const float* z2b2_c2 = (const float*)d_in[48];
    const float* z2b2_rb = (const float*)d_in[49];
    const float* z2b2_rw = (const float*)d_in[50];

    // Byte arena (~115 MB): halo-padded f16 activations + packed f16 weights.
    char* base = (char*)d_ws;
    size_t off = 0;
    auto take = [&](size_t bytes) {
        off = (off + 63) & ~(size_t)63;
        void* p = base + off; off += bytes; return p;
    };
    auto takeH = [&](size_t n) { return (h16*)take(n * sizeof(h16)); };
    auto takeF = [&](size_t n) { return (float*)take(n * sizeof(float)); };

    float* ENC_TH = takeF(64*512);
    float* QT     = takeF(64*256);
    // padded f16 activation slabs (counts in halves):
    h16* S0 = takeH(64LL*130*256);   // h [64,130,256]            / z1_in
    h16* S1 = takeH(64LL*34*3584);   // mid scratch (max shape [64,34,3584])
    h16* S2 = takeH(64LL*130*512);   // eb1_out                   / wconv_out (ew2)
    h16* S3 = takeH(64LL*130*512);   // w = eb2_out               / z1_out
    h16* S4 = takeH(64LL*130*512);   // ew / z2c_out              / z2_back
    h16* S5 = takeH(64LL*18*3584);   // z2_in [64,130,256]        / z2b1_out [64,18,3584]
    h16* S6 = takeH(64LL*18*3584);   // z2_roi [64,18,3584]       / ct_out [64,34,1792]
    h16* S7 = takeH(64LL*34*3584);   // z2b2_out
    h16* Z1M = takeH(64LL*128*128);  // unpadded
    h16* Z2M = takeH(64LL*128*128);
    h16* QZ  = takeH(64LL*130*256);  // padded [64,130,256]
    h16* D0  = takeH(64LL*258*256);  // up1 [64,258,256] / up2 [64,514,128]
    h16* D1  = takeH(64LL*258*128);  // [64,258,128] / [64,514,64]
    h16* D2  = takeH(64LL*258*128);
    // packed f16 weights:
    h16* P_eb1c1 = takeH(98304);
    h16* P_eb1c2 = takeH(196608);
    h16* P_eb1rw = takeH(32768);
    h16* P_eb2c1 = takeH(196608);
    h16* P_eb2c2 = takeH(196608);
    h16* P_wc1   = takeH(196608);
    h16* P_wc2   = takeH(196608);
    h16* P_z1c1  = takeH(98304);
    h16* P_z1c2  = takeH(196608);
    h16* P_z1rw  = takeH(32768);
    h16* P_z2c1  = takeH(98304);
    h16* P_z2c2  = takeH(196608);
    h16* P_z2rw  = takeH(32768);
    h16* P_b1c1  = takeH(1376256);
    h16* P_b1c2  = takeH(1376256);
    h16* P_b2c1  = takeH(688128);
    h16* P_b2c2  = takeH(1376256);
    h16* P_b2rw  = takeH(229376);
    h16* P_d1w1  = takeH(98304);
    h16* P_d1w2  = takeH(49152);
    h16* P_d2w1  = takeH(24576);
    h16* P_d2w2  = takeH(12288);

    // ---- pack all conv weights into WMMA fragment order ----
    auto pack = [&](const float* w, h16* p, int CinPG, int taps, int total) {
        pack_w_k<<<g1(total), 256, 0, stream>>>(w, p, CinPG, taps, total);
    };
    pack(eb1_c1, P_eb1c1,  64, 3, 98304);
    pack(eb1_c2, P_eb1c2, 128, 3, 196608);
    pack(eb1_rw, P_eb1rw,  64, 1, 32768);
    pack(eb2_c1, P_eb2c1, 128, 3, 196608);
    pack(eb2_c2, P_eb2c2, 128, 3, 196608);
    pack(wc_c1,  P_wc1,   128, 3, 196608);
    pack(wc_c2,  P_wc2,   128, 3, 196608);
    pack(z1_c1,  P_z1c1,   64, 3, 98304);
    pack(z1_c2,  P_z1c2,  128, 3, 196608);
    pack(z1_rw,  P_z1rw,   64, 1, 32768);
    pack(z2_c1,  P_z2c1,   64, 3, 98304);
    pack(z2_c2,  P_z2c2,  128, 3, 196608);
    pack(z2_rw,  P_z2rw,   64, 1, 32768);
    pack(z2b1_c1,P_b1c1,  128, 3, 1376256);
    pack(z2b1_c2,P_b1c2,  128, 3, 1376256);
    pack(z2b2_c1,P_b2c1,   64, 3, 688128);
    pack(z2b2_c2,P_b2c2,  128, 3, 1376256);
    pack(z2b2_rw,P_b2rw,   64, 1, 229376);
    pack(dc1_w1, P_d1w1,  256, 3, 98304);
    pack(dc1_w2, P_d1w2,  128, 3, 49152);
    pack(dc2_w1, P_d2w1,  128, 3, 24576);
    pack(dc2_w2, P_d2w2,   64, 3, 12288);

    const h16* NH = nullptr; const float* NF = nullptr;

    // ---- encoder ----
    enc_conv7_pool_t<<<g1(64LL*128*256), 256, 0, stream>>>(x, enc_c1, S0);
    launch_conv(S0, P_eb1c1, NH, NH, NF, NF, NF, NF, S1,  64,   0, 128, 128, 4, 0, 1, stream);
    launch_conv(S1, P_eb1c2, S0, P_eb1rw, eb1_rb, NF, NF, NF, S2, 128, 64, 128, 128, 4, 2, 1, stream);
    launch_conv(S2, P_eb2c1, NH, NH, NF, NF, NF, NF, S1, 128,   0, 128, 128, 4, 0, 1, stream);
    launch_conv(S1, P_eb2c2, S2, NH, NF, NF, NF, NF, S3, 128, 128, 128, 128, 4, 1, 1, stream);

    // ---- theta conditioning ----
    theta_mlp<<<g1(256LL*128), 256, 0, stream>>>(ithetas, mlp1_w, mlp1_b, ENC_TH, 256, 128);
    theta_mlp<<<g1(64LL*256),  256, 0, stream>>>(qtheta,  mlp2_w, mlp2_b, QT,     64, 256);
    ew_mul_t<<<g1(64LL*128*512), 256, 0, stream>>>(ENC_TH, S3, S4);
    launch_conv(S4, P_wc1, NH, NH, NF, NF, NF, NF, S1, 128,   0, 128, 128, 4, 0, 1, stream);
    launch_conv(S1, P_wc2, S4, NH, NF, NF, NF, NF, S2, 128, 128, 128, 128, 4, 1, 1, stream);
    split_t<<<g1(64LL*128*256), 256, 0, stream>>>(S2, S0, S5);

    // ---- z1 / z2 branches ----
    launch_conv(S0, P_z1c1, NH, NH, NF, NF, NF, NF, S1,  64,  0, 128, 128, 4, 0, 1, stream);
    launch_conv(S1, P_z1c2, S0, P_z1rw, z1_rb, NF, NF, NF, S3, 128, 64, 128, 128, 4, 2, 1, stream);
    launch_conv(S5, P_z2c1, NH, NH, NF, NF, NF, NF, S1,  64,  0, 128, 128, 4, 0, 1, stream);
    launch_conv(S1, P_z2c2, S5, P_z2rw, z2_rb, NF, NF, NF, S4, 128, 64, 128, 128, 4, 2, 1, stream);

    // ---- ROI processing ----
    roi_align_t<<<g1(64LL*16*3584), 256, 0, stream>>>(S4, rois, S6);
    launch_conv(S6, P_b1c1, NH, NH, NF, NF, NF, NF, S1, 128,   0, 128, 16, 28, 0, 1, stream);
    launch_conv(S1, P_b1c2, S6, NH, NF, NF, NF, NF, S5, 128, 128, 128, 16, 28, 1, 1, stream);
    convT_t<<<g1(64LL*32*1792), 256, 0, stream>>>(S5, ct_w, ct_b, S6);
    launch_conv(S6, P_b2c1, NH, NH, NF, NF, NF, NF, S1,  64,  0, 128, 32, 28, 0, 1, stream);
    launch_conv(S1, P_b2c2, S6, P_b2rw, z2b2_rb, NF, NF, NF, S7, 128, 64, 128, 32, 28, 2, 1, stream);
    roi_rev_t<<<g1(64LL*128*512), 256, 0, stream>>>(S7, rois, S4);

    // ---- lead mixing ----
    lead_mean_t<<<g1(64LL*128*128), 256, 0, stream>>>(S3, Z1M);
    lead_mean_t<<<g1(64LL*128*128), 256, 0, stream>>>(S4, Z2M);

    // ---- 3 decoder passes (latent, sh_p, sh_l) ----
    float* OUT = (float*)d_out;
    for (int mode = 0; mode < 3; ++mode) {
        build_qz_t<<<g1(64LL*128*256), 256, 0, stream>>>(S3, S4, Z1M, Z2M, QT, QZ, mode);
        up2_t<<<g1(64LL*256*256), 256, 0, stream>>>(QZ, D0, 256, 128);
        launch_conv(D0, P_d1w1, NH, NH, NF, dc1_g1, dc1_b1, dc1_be1, D1, 256, 0, 128, 256, 1, 0, 1, stream);
        launch_conv(D1, P_d1w2, NH, NH, NF, dc1_g2, dc1_b2, dc1_be2, D2, 128, 0, 128, 256, 1, 0, 1, stream);
        up2_t<<<g1(64LL*128*512), 256, 0, stream>>>(D2, D0, 128, 256);
        launch_conv(D0, P_d2w1, NH, NH, NF, dc2_g1, dc2_b1, dc2_be1, D1, 128, 0, 64, 512, 1, 0, 1, stream);
        launch_conv(D1, P_d2w2, NH, NH, NF, dc2_g2, dc2_b2, dc2_be2, D2,  64, 0, 64, 512, 1, 0, 1, stream);
        halo_zero<<<g1(64LL*2*64), 256, 0, stream>>>(D2, 512, 64);
        outconv_sig_t<<<g1(64LL*512), 256, 0, stream>>>(D2, out_w, out_b, OUT + (size_t)mode*64*512);
    }
}
